// Classifier_new_38568806318158
// MI455X (gfx1250) — matrix operand order repaired and checked by reading, not compiled, not run on hardware
//
#include <hip/hip_runtime.h>
#include <hip/hip_bf16.h>

// ---------------- CDNA5 WMMA plumbing ----------------
typedef __attribute__((ext_vector_type(16))) __bf16 v16bf;
typedef __attribute__((ext_vector_type(8)))  float  v8f;
typedef __attribute__((ext_vector_type(4)))  int    v4i_t;
typedef __attribute__((ext_vector_type(8)))  int    v8i_t;

union BFfrag {
    v16bf v;
    unsigned short s[16];
    unsigned u[8];
    uint4 q[2];
};

static __device__ inline unsigned short f2bf(float f) {
    unsigned u = __builtin_bit_cast(unsigned, f);
    return (unsigned short)(u >> 16);
}
static __device__ inline unsigned pack2bf(float a, float b) {
    unsigned ua = __builtin_bit_cast(unsigned, a);
    unsigned ub = __builtin_bit_cast(unsigned, b);
    return (ua >> 16) | (ub & 0xffff0000u);
}
static __device__ inline void pack8(BFfrag& f, int u0, float4 a, float4 b) {
    f.u[u0 + 0] = pack2bf(a.x, a.y);
    f.u[u0 + 1] = pack2bf(a.z, a.w);
    f.u[u0 + 2] = pack2bf(b.x, b.y);
    f.u[u0 + 3] = pack2bf(b.z, b.w);
}
static __device__ inline v8f wmma_bf(const BFfrag& a, const BFfrag& b, v8f c) {
    return __builtin_amdgcn_wmma_f32_16x16x32_bf16(false, a.v, false, b.v,
                                                   (short)0, c, false, false);
}

// ---------------- model constants ----------------
#define BATCH   32
#define S_LEN   257
#define SPAD    272
#define DMODEL  128
#define INNER_  256
#define NHEAD   4
#define DHEAD   64
#define EPSF    1e-5f

static __device__ float redsum(float v, float* sh, int nt) {
    sh[threadIdx.x] = v; __syncthreads();
    for (int o = nt >> 1; o > 0; o >>= 1) {
        if ((int)threadIdx.x < o) sh[threadIdx.x] += sh[threadIdx.x + o];
        __syncthreads();
    }
    float r = sh[0]; __syncthreads();
    return r;
}

// ---------------- NCHW -> NHWC ----------------
__global__ void k_nchw2nhwc(const float* __restrict__ in, float* __restrict__ out,
                            int C, int Hh, int Wn) {
    size_t idx = (size_t)blockIdx.x * blockDim.x + threadIdx.x;
    size_t tot = (size_t)BATCH * C * Hh * Wn;
    if (idx >= tot) return;
    int w = idx % Wn; size_t r = idx / Wn;
    int h = r % Hh; r /= Hh;
    int c = r % C;  int b = r / C;
    out[(((size_t)b * Hh + h) * Wn + w) * C + c] = in[idx];
}

// ---------------- weight repack: (K x N) f32 -> bf16 B-fragment order ----------------
__global__ void k_packW(const float* __restrict__ W, unsigned short* __restrict__ P,
                        int K, int N) {
    int idx = blockIdx.x * blockDim.x + threadIdx.x;
    if (idx >= K * N) return;
    int kk = idx & 31; int rest = idx >> 5;
    int n = rest % N;  int kt = rest / N;
    P[idx] = f2bf(W[(size_t)(kt * 32 + kk) * N + n]);
}

// conv weights (Cout,Cin,1,3) -> per-tap B-fragment order, zero-padded
__global__ void k_packConvW(const float* __restrict__ W, unsigned short* __restrict__ P,
                            int Cin, int Cout, int KT, int Np) {
    int idx = blockIdx.x * blockDim.x + threadIdx.x;
    int tot = 3 * KT * Np * 32;
    if (idx >= tot) return;
    int kk = idx & 31; int rest = idx >> 5;
    int n = rest % Np; rest /= Np;
    int kt = rest % KT; int t = rest / KT;
    int k = kt * 32 + kk;
    float v = (n < Cout && k < Cin) ? W[((size_t)n * Cin + k) * 3 + t] : 0.f;
    P[idx] = f2bf(v);
}

// ---------------- conv 1x3 implicit GEMM (channels-last, bf16 WMMA, packed W) ----------------
__global__ void k_conv1x3(const float* __restrict__ in, const unsigned short* __restrict__ Pw,
                          const float* __restrict__ bias, float* __restrict__ out,
                          int Cin, int Cout, int Hh, int Win, int Wout,
                          int stride, int pad, int KT, int Np) {
    int lane = threadIdx.x;
    int lm = lane & 15, lg = lane >> 4;
    int tilesN = Np >> 4;
    int tilesM = (Hh * Wout) >> 4;
    int bidx = blockIdx.x;
    int tn = bidx % tilesN; bidx /= tilesN;
    int tm = bidx % tilesM; int bb = bidx / tilesM;

    int m  = tm * 16 + lm;
    int h  = m / Wout, wo = m % Wout;
    int n  = tn * 16 + lm;

    v8f acc = {};
    for (int t = 0; t < 3; ++t) {
        int wi = wo * stride + t - pad;
        bool valid = (wi >= 0) && (wi < Win);
        const float* arow = in + (((size_t)bb * Hh + h) * Win + (valid ? wi : 0)) * Cin;
        for (int kt = 0; kt < KT; ++kt) {
            int ka = kt * 32 + lg * 8;
            BFfrag af, bfm;
            if (Cin >= 32) {           // full K-chunks: vector loads
                if (valid) {
                    float4 a0 = *(const float4*)(arow + ka);
                    float4 a1 = *(const float4*)(arow + ka + 4);
                    float4 a2 = *(const float4*)(arow + ka + 16);
                    float4 a3 = *(const float4*)(arow + ka + 20);
                    pack8(af, 0, a0, a1);
                    pack8(af, 4, a2, a3);
                } else {
#pragma unroll
                    for (int i = 0; i < 8; ++i) af.u[i] = 0u;
                }
            } else {                   // Cin==3: only lanes 0-15 carry K=0..2
#pragma unroll
                for (int i = 0; i < 8; ++i) af.u[i] = 0u;
                if (valid && lg == 0) {
                    float a0 = arow[0], a1 = arow[1], a2 = arow[2];
                    af.u[0] = pack2bf(a0, a1);
                    af.u[1] = pack2bf(a2, 0.f);
                }
            }
            const uint4* bp = (const uint4*)(Pw + ((((size_t)t * KT + kt) * Np + n) << 5) + lg * 8);
            bfm.q[0] = bp[0];
            bfm.q[1] = bp[2];
            acc = wmma_bf(af, bfm, acc);
        }
    }
    if (n < Cout) {
        float bv = bias[n];
#pragma unroll
        for (int r = 0; r < 8; ++r) {
            int mr = tm * 16 + r + 8 * lg;
            int h2 = mr / Wout, w2 = mr % Wout;
            out[(((size_t)bb * Hh + h2) * Wout + w2) * Cout + n] = acc[r] + bv;
        }
    }
}

// ---------------- instance-norm: coalesced stats (atomics) + coalesced apply ----------------
__global__ void k_zero(float* __restrict__ p, int n) {
    int i = blockIdx.x * blockDim.x + threadIdx.x;
    if (i < n) p[i] = 0.f;
}

#define IN_CHUNK 16384
__global__ void k_instats(const float* __restrict__ x, float* __restrict__ st,
                          int C, int HW) {
    size_t plane = (size_t)HW * C;
    int b = blockIdx.y;
    const float* xp = x + (size_t)b * plane;
    size_t start = (size_t)blockIdx.x * IN_CHUNK;
    float s1 = 0.f, s2 = 0.f;
#pragma unroll 4
    for (int it = 0; it < IN_CHUNK / 256; ++it) {
        size_t o = start + (size_t)it * 256 + threadIdx.x;
        if (o < plane) { float v = xp[o]; s1 += v; s2 += v * v; }
    }
    __shared__ float sh[256];
    int tid = threadIdx.x;
    sh[tid] = s1; __syncthreads();
    for (int off = 128; off >= C; off >>= 1) {
        if (tid < off) sh[tid] += sh[tid + off];
        __syncthreads();
    }
    float r1 = (tid < C) ? sh[tid] : 0.f; __syncthreads();
    sh[tid] = s2; __syncthreads();
    for (int off = 128; off >= C; off >>= 1) {
        if (tid < off) sh[tid] += sh[tid + off];
        __syncthreads();
    }
    if (tid < C) {
        atomicAdd(&st[((size_t)b * C + tid) * 2],     r1);
        atomicAdd(&st[((size_t)b * C + tid) * 2 + 1], sh[tid]);
    }
}

__global__ void k_inapply(float* __restrict__ x, const float* __restrict__ st,
                          const float* __restrict__ g, const float* __restrict__ bta,
                          const float* __restrict__ al, int C, int HW) {
    size_t idx = (size_t)blockIdx.x * blockDim.x + threadIdx.x;
    size_t plane = (size_t)HW * C;
    size_t tot = (size_t)BATCH * plane;
    if (idx >= tot) return;
    int b = (int)(idx / plane);
    int c = (int)(idx % (size_t)C);
    float s1 = st[((size_t)b * C + c) * 2];
    float s2 = st[((size_t)b * C + c) * 2 + 1];
    float mean = s1 / HW;
    float var  = s2 / HW - mean * mean;
    float rstd = rsqrtf(var + EPSF);
    float v = x[idx];
    v = (v - mean) * rstd * g[c] + bta[c];
    x[idx] = (v >= 0.f) ? v : al[c] * v;
}

// ---------------- build sequence ----------------
__global__ void k_seq(const float* __restrict__ conv, const float* __restrict__ cls,
                      float* __restrict__ seq) {
    int idx = blockIdx.x * blockDim.x + threadIdx.x;
    int tot = BATCH * S_LEN * DMODEL;
    if (idx >= tot) return;
    int d = idx % DMODEL; int r = idx / DMODEL;
    int s = r % S_LEN;    int b = r / S_LEN;
    seq[idx] = (s < 256) ? conv[((size_t)b * 256 + s) * DMODEL + d] : cls[d];
}

// ---------------- layernorm ----------------
__global__ void k_ln(const float* __restrict__ in, const float* __restrict__ g,
                     const float* __restrict__ bta, float* __restrict__ out) {
    int tok = blockIdx.x, d = threadIdx.x;
    __shared__ float sh[DMODEL];
    float v = in[(size_t)tok * DMODEL + d];
    float mu  = redsum(v, sh, DMODEL) / DMODEL;
    float dv  = v - mu;
    float var = redsum(dv * dv, sh, DMODEL) / DMODEL;
    out[(size_t)tok * DMODEL + d] = dv * rsqrtf(var + EPSF) * g[d] + bta[d];
}

// ---------------- GEMM: 4 waves / block, 16x64 tile, A panel staged in LDS via TDM ----------
// out = A(MxK) @ W(KxN) [+bias] [+res]; W pre-packed bf16 fragment order.
#define GEMM_MAXK 256
__global__ void k_gemm(const float* __restrict__ A, const unsigned short* __restrict__ P,
                       const float* __restrict__ bias, const float* __restrict__ res,
                       float* __restrict__ out, int K, int N) {
    __shared__ float ldsA[16 * GEMM_MAXK];
    int tid = threadIdx.x;
    int lane = tid & 31;
    int lm = lane & 15, lg = lane >> 4;
    int tiles64 = N >> 6;
    int tn = blockIdx.x % tiles64;
    int tm = blockIdx.x / tiles64;

    // --- Tensor Data Mover: wave wv DMAs A rows [tm*16+wv*4, +4) (contiguous 4*K f32) ---
    {
        int wv = __builtin_amdgcn_readfirstlane(tid >> 5);     // wave-uniform, SGPR
        unsigned L = 4u * (unsigned)K;                          // elements (<= 1024)
        unsigned ldsoff = (unsigned)(size_t)(&ldsA[0]) + (unsigned)wv * L * 4u;
        unsigned long long ga =
            (unsigned long long)(size_t)(A + ((size_t)tm * 16 + (size_t)wv * 4) * K);
        v4i_t g0;
        g0[0] = 1;                                              // count=1 (valid D#)
        g0[1] = (int)ldsoff;                                    // lds_addr (bytes)
        g0[2] = (int)(unsigned)ga;                              // global_addr[31:0]
        g0[3] = (int)(((unsigned)(ga >> 32) & 0x01FFFFFFu) | 0x80000000u); // addr[56:32] | type=2
        v8i_t g1;
        g1[0] = 0x20000;                                        // data_size=4B, mask=0
        g1[1] = (int)((L & 0xffffu) << 16);                     // tensor_dim0 lo16
        g1[2] = (int)(((L >> 16) & 0xffffu) | (1u << 16));      // tensor_dim0 hi | tensor_dim1=1
        g1[3] = (int)((L & 0xffffu) << 16);                     // tile_dim0 = L
        g1[4] = 1;                                              // tile_dim1 = 1
        g1[5] = (int)L;                                         // tensor_dim0_stride = L
        g1[6] = 0;
        g1[7] = 0;
        asm volatile("tensor_load_to_lds %0, %1" :: "s"(g0), "s"(g1) : "memory");
        __builtin_amdgcn_s_wait_tensorcnt(0);
    }
    __syncthreads();

    int wv = tid >> 5;
    int col = tn * 64 + wv * 16 + lm;
    v8f acc = {};
    for (int k0 = 0; k0 < K; k0 += 32) {
        int ka = k0 + lg * 8;
        const float* ar = &ldsA[lm * K + ka];
        float4 a0 = *(const float4*)(ar);          // ds_load_b128
        float4 a1 = *(const float4*)(ar + 4);
        float4 a2 = *(const float4*)(ar + 16);
        float4 a3 = *(const float4*)(ar + 20);
        BFfrag af, bfm;
        pack8(af, 0, a0, a1);
        pack8(af, 4, a2, a3);
        const uint4* bp = (const uint4*)(P + ((size_t)(k0 >> 5) * N + col) * 32 + lg * 8);
        bfm.q[0] = bp[0];
        bfm.q[1] = bp[2];
        acc = wmma_bf(af, bfm, acc);
    }
#pragma unroll
    for (int r = 0; r < 8; ++r) {
        size_t row = (size_t)tm * 16 + r + 8 * lg;
        size_t o = row * N + col;
        float v = acc[r];
        if (bias) v += bias[col];
        if (res)  v += res[o];
        out[o] = v;
    }
}

// ---------------- causal depthwise conv (K=4) + SiLU ----------------
__global__ void k_dwconv(const float* __restrict__ up, const float* __restrict__ cw,
                         const float* __restrict__ cb, float* __restrict__ ca, int total) {
    int idx = blockIdx.x * blockDim.x + threadIdx.x;
    if (idx >= total) return;
    int c = idx % INNER_; int r = idx / INNER_;
    int s = r % S_LEN;    int b = r / S_LEN;
    float acc = 0.f;
#pragma unroll
    for (int j = 0; j < 4; ++j) {
        int sp = s - 3 + j;
        if (sp >= 0) acc += cw[c * 4 + j] * up[((size_t)b * S_LEN + sp) * 512 + c];
    }
    acc += cb[c];
    ca[idx] = acc / (1.f + __expf(-acc));
}

// ---------------- block-diagonal 4x4 linear ----------------
__global__ void k_blocked(const float* __restrict__ in, int ld,
                          const float* __restrict__ w, float* __restrict__ out, int total) {
    int idx = blockIdx.x * blockDim.x + threadIdx.x;
    if (idx >= total) return;
    int nb = idx & 63; int tok = idx >> 6;
    float4 xv = *(const float4*)(in + (size_t)tok * ld + nb * 4);
    const float4* wp = (const float4*)(w + nb * 16);
    float4 o;
    float4 w0 = wp[0], w1 = wp[1], w2 = wp[2], w3 = wp[3];
    o.x = w0.x * xv.x + w0.y * xv.y + w0.z * xv.z + w0.w * xv.w;
    o.y = w1.x * xv.x + w1.y * xv.y + w1.z * xv.z + w1.w * xv.w;
    o.z = w2.x * xv.x + w2.y * xv.y + w2.z * xv.z + w2.w * xv.w;
    o.w = w3.x * xv.x + w3.y * xv.y + w3.z * xv.z + w3.w * xv.w;
    *(float4*)(out + (size_t)tok * INNER_ + nb * 4) = o;
}

// ---------------- gates ----------------
__global__ void k_gates(const float* __restrict__ q, const float* __restrict__ k,
                        const float* __restrict__ v, const float* __restrict__ wi,
                        const float* __restrict__ bi, const float* __restrict__ wf,
                        const float* __restrict__ bff, float* __restrict__ ig,
                        float* __restrict__ lf, int total) {
    int idx = blockIdx.x * blockDim.x + threadIdx.x;
    if (idx >= total) return;
    int hh = idx & 3; int tok = idx >> 2;
    const float* qp = q + (size_t)tok * INNER_;
    const float* kp = k + (size_t)tok * INNER_;
    const float* vp = v + (size_t)tok * INNER_;
    float ai = 0.f, af = 0.f;
    for (int j = 0; j < INNER_; ++j) { ai += qp[j] * wi[j * 4 + hh];         af += qp[j] * wf[j * 4 + hh]; }
    for (int j = 0; j < INNER_; ++j) { ai += kp[j] * wi[(256 + j) * 4 + hh]; af += kp[j] * wf[(256 + j) * 4 + hh]; }
    for (int j = 0; j < INNER_; ++j) { ai += vp[j] * wi[(512 + j) * 4 + hh]; af += vp[j] * wf[(512 + j) * 4 + hh]; }
    ai += bi[hh]; af += bff[hh];
    ig[idx] = ai;
    lf[idx] = (af < 0.f) ? (af - log1pf(__expf(af))) : (-log1pf(__expf(-af)));
}

// ---------------- per (b,h) scans ----------------
__global__ void k_scan(const float* __restrict__ ig, const float* __restrict__ lf,
                       float* __restrict__ cf, float* __restrict__ ub, float* __restrict__ ab) {
    int idx = blockIdx.x * blockDim.x + threadIdx.x;
    if (idx >= BATCH * NHEAD) return;
    int hh = idx & 3, b = idx >> 2;
    float c = 0.f, am = -3e38f;
    for (int s = 0; s < S_LEN; ++s) {
        size_t gi = ((size_t)b * S_LEN + s) * NHEAD + hh;
        c += lf[gi];
        float u = ig[gi] - c;
        am = fmaxf(am, u);
        size_t o = ((size_t)(b * NHEAD + hh)) * S_LEN + s;
        cf[o] = c; ub[o] = u; ab[o] = am;
    }
}

// ---------------- V transpose: vt[(b,h,d), jpad] with zero pad ----------------
__global__ void k_vtrans(const float* __restrict__ v, float* __restrict__ vt) {
    int idx = blockIdx.x * blockDim.x + threadIdx.x;
    int tot = BATCH * NHEAD * DHEAD * SPAD;
    if (idx >= tot) return;
    int j = idx % SPAD; int rest = idx / SPAD;
    int d = rest % DHEAD; rest /= DHEAD;
    int hh = rest % NHEAD; int b = rest / NHEAD;
    vt[idx] = (j < S_LEN) ? v[((size_t)b * S_LEN + j) * INNER_ + hh * DHEAD + d] : 0.f;
}

// ---------------- mLSTM attention core ----------------
__global__ void k_attn(const float* __restrict__ q, const float* __restrict__ kk,
                       const float* __restrict__ vt, const float* __restrict__ cf,
                       const float* __restrict__ ub, const float* __restrict__ ab,
                       float* __restrict__ hb) {
    int bid = blockIdx.x;
    int tile = bid % 17; bid /= 17;
    int hh = bid % NHEAD; int b = bid / NHEAD;
    int lane = threadIdx.x;
    int lm = lane & 15, lg = lane >> 4;
    int i0 = tile * 16;
    size_t bh = (size_t)(b * NHEAD + hh) * S_LEN;

    float aival[8], cfval[8];
#pragma unroll
    for (int r = 0; r < 8; ++r) {
        int ir = i0 + r + 8 * lg;
        if (ir < S_LEN) { aival[r] = ab[bh + ir]; cfval[r] = cf[bh + ir]; }
        else            { aival[r] = 0.f;         cfval[r] = 0.f; }
    }

    int irow = i0 + lm; if (irow >= S_LEN) irow = S_LEN - 1;
    const float4* qp = (const float4*)(q + ((size_t)b * S_LEN + irow) * INNER_ + hh * DHEAD);
    BFfrag qa0, qa1;
    pack8(qa0, 0, qp[lg * 2],      qp[lg * 2 + 1]);
    pack8(qa0, 4, qp[lg * 2 + 4],  qp[lg * 2 + 5]);
    pack8(qa1, 0, qp[lg * 2 + 8],  qp[lg * 2 + 9]);
    pack8(qa1, 4, qp[lg * 2 + 12], qp[lg * 2 + 13]);

    float rs[8] = {};
    v8f hacc[4] = {};
    __shared__ unsigned short cm[256];

    for (int j0 = 0; j0 <= i0; j0 += 16) {
        int jc = j0 + lm; if (jc >= S_LEN) jc = S_LEN - 1;
        const float4* kp = (const float4*)(kk + ((size_t)b * S_LEN + jc) * INNER_ + hh * DHEAD);
        const float kscale = 0.125f;  // 1/sqrt(DH)
        BFfrag kb0, kb1;
#pragma unroll
        for (int half = 0; half < 2; ++half) {
            BFfrag& kbf = half ? kb1 : kb0;
            float4 c0 = kp[half * 8 + lg * 2];
            float4 c1 = kp[half * 8 + lg * 2 + 1];
            float4 c2 = kp[half * 8 + lg * 2 + 4];
            float4 c3 = kp[half * 8 + lg * 2 + 5];
            c0.x *= kscale; c0.y *= kscale; c0.z *= kscale; c0.w *= kscale;
            c1.x *= kscale; c1.y *= kscale; c1.z *= kscale; c1.w *= kscale;
            c2.x *= kscale; c2.y *= kscale; c2.z *= kscale; c2.w *= kscale;
            c3.x *= kscale; c3.y *= kscale; c3.z *= kscale; c3.w *= kscale;
            pack8(kbf, 0, c0, c1);
            pack8(kbf, 4, c2, c3);
        }
        v8f sc = {};
        sc = wmma_bf(qa0, kb0, sc);
        sc = wmma_bf(qa1, kb1, sc);

        int jcol = j0 + lm;
        float uj = (jcol < S_LEN) ? ub[bh + jcol] : -1e30f;
        __syncthreads();
#pragma unroll
        for (int r = 0; r < 8; ++r) {
            int ir = i0 + r + 8 * lg;
            float w = (jcol <= ir && jcol < S_LEN && ir < S_LEN)
                    ? __expf(uj - aival[r]) : 0.f;
            float cmv = sc[r] * w;
            rs[r] += cmv;
            cm[(r + 8 * lg) * 16 + lm] = f2bf(cmv);
        }
        __syncthreads();

        BFfrag pa;
        pa.q[0] = lg ? make_uint4(0u, 0u, 0u, 0u) : *(const uint4*)(&cm[lm * 16]);   // ds_load_b128
        pa.q[1] = lg ? make_uint4(0u, 0u, 0u, 0u) : *(const uint4*)(&cm[lm * 16 + 8]);
#pragma unroll
        for (int dt = 0; dt < 4; ++dt) {
            int dcol = dt * 16 + lm;
            BFfrag pb;
            if (lg == 0) {
                const float4* vp = (const float4*)(vt +
                    ((size_t)((b * NHEAD + hh) * DHEAD + dcol)) * SPAD + j0);
                pack8(pb, 0, vp[0], vp[1]);
                pack8(pb, 4, vp[2], vp[3]);
            } else {
#pragma unroll
                for (int i = 0; i < 8; ++i) pb.u[i] = 0u;
            }
            hacc[dt] = wmma_bf(pa, pb, hacc[dt]);
        }
    }

#pragma unroll
    for (int r = 0; r < 8; ++r) {
        float x = rs[r];
        x += __shfl_xor(x, 1); x += __shfl_xor(x, 2);
        x += __shfl_xor(x, 4); x += __shfl_xor(x, 8);
        rs[r] = x;
    }
#pragma unroll
    for (int r = 0; r < 8; ++r) {
        int ir = i0 + r + 8 * lg;
        if (ir >= S_LEN) continue;
        float nr = fmaxf(fabsf(rs[r]), __expf(-(aival[r] + cfval[r])));
        float inv = 1.f / nr;
#pragma unroll
        for (int dt = 0; dt < 4; ++dt)
            hb[((size_t)b * S_LEN + ir) * INNER_ + hh * DHEAD + dt * 16 + lm] =
                hacc[dt][r] * inv;
    }
}

// ---------------- group-norm + skip + SiLU(z) gating ----------------
__global__ void k_gnfuse(const float* __restrict__ hbuf, const float* __restrict__ up,
                         const float* __restrict__ ca, const float* __restrict__ gng,
                         const float* __restrict__ skip, float* __restrict__ gin) {
    int bid = blockIdx.x;
    int hh = bid % NHEAD; int tok = bid / NHEAD;
    int d = threadIdx.x, c = hh * DHEAD + d;
    __shared__ float sh[DHEAD];
    float hv = hbuf[(size_t)tok * INNER_ + c];
    float mu  = redsum(hv, sh, DHEAD) / DHEAD;
    float dv  = hv - mu;
    float var = redsum(dv * dv, sh, DHEAD) / DHEAD;
    float val = dv * rsqrtf(var + EPSF) * gng[c];
    val += skip[c] * ca[(size_t)tok * INNER_ + c];
    float z = up[(size_t)tok * 512 + 256 + c];
    gin[(size_t)tok * INNER_ + c] = val * (z / (1.f + __expf(-z)));
}

// ---------------- head ----------------
__global__ void k_head(const float* __restrict__ x, const float* __restrict__ pg,
                       const float* __restrict__ pb, const float* __restrict__ lg_,
                       const float* __restrict__ lb_, const float* __restrict__ fw,
                       const float* __restrict__ fb, float* __restrict__ out) {
    int b = blockIdx.x, d = threadIdx.x;
    __shared__ float sh[DMODEL];
    __shared__ float ybuf[DMODEL];
    float v = x[((size_t)b * S_LEN + 256) * DMODEL + d];
    float mu  = redsum(v, sh, DMODEL) / DMODEL;
    float dv  = v - mu;
    float var = redsum(dv * dv, sh, DMODEL) / DMODEL;
    float h1 = dv * rsqrtf(var + EPSF) * pg[d] + pb[d];
    mu  = redsum(h1, sh, DMODEL) / DMODEL;
    dv  = h1 - mu;
    var = redsum(dv * dv, sh, DMODEL) / DMODEL;
    float y = dv * rsqrtf(var + EPSF) * lg_[d] + lb_[d];
    ybuf[d] = fmaxf(y, 0.f);
    __syncthreads();
    if (d < 10) {
        float acc = fb[d];
        for (int j = 0; j < DMODEL; ++j) acc += ybuf[j] * fw[j * 10 + d];
        out[b * 10 + d] = acc;
    }
}

// ================== host side ==================
extern "C" void kernel_launch(void* const* d_in, const int* in_sizes, int n_in,
                              void* d_out, int out_size, void* d_ws, size_t ws_size,
                              hipStream_t stream) {
    (void)in_sizes; (void)n_in; (void)out_size; (void)ws_size;
    const float* x_in = (const float*)d_in[0];
    const float* cls  = (const float*)d_in[31];
    const float* post_g = (const float*)d_in[92];
    const float* post_b = (const float*)d_in[93];
    const float* fc_lg  = (const float*)d_in[94];
    const float* fc_lb  = (const float*)d_in[95];
    const float* fc_w   = (const float*)d_in[96];
    const float* fc_b   = (const float*)d_in[97];

    float* ws = (float*)d_ws;
    const size_t SZ_A = 67108864ULL, SZ_B = 134217728ULL;
    float* bufA = ws;
    float* bufB = ws + SZ_A;
    size_t off = SZ_A + SZ_B;
    const size_t TOK = (size_t)BATCH * S_LEN;
    float* xb   = ws + off; off += TOK * DMODEL;
    float* xn   = ws + off; off += TOK * DMODEL;
    float* up   = ws + off; off += TOK * 512;
    float* ca   = ws + off; off += TOK * INNER_;
    float* qb   = ws + off; off += TOK * INNER_;
    float* kb   = ws + off; off += TOK * INNER_;
    float* vb   = ws + off; off += TOK * INNER_;
    float* igb  = ws + off; off += TOK * NHEAD;
    float* lfb  = ws + off; off += TOK * NHEAD;
    float* cfb  = ws + off; off += TOK * NHEAD;
    float* ubuf = ws + off; off += TOK * NHEAD;
    float* abuf = ws + off; off += TOK * NHEAD;
    float* hbuf = ws + off; off += TOK * INNER_;
    float* gin  = ws + off; off += TOK * INNER_;
    float* vt   = ws + off; off += (size_t)BATCH * NHEAD * DHEAD * SPAD;
    float* stat = ws + off; off += (size_t)BATCH * 128 * 2;
    unsigned short* packG = (unsigned short*)(ws + off); off += 32768;
    unsigned short* packC = (unsigned short*)(ws + off); off += 32768;

    // ---- conv stack (channels-last) ----
    {
        size_t tot = (size_t)BATCH * 3 * 256 * 511;
        k_nchw2nhwc<<<(unsigned)((tot + 255) / 256), 256, 0, stream>>>(x_in, bufA, 3, 256, 511);
    }
    const int Cin_[6]  = {3, 32, 64, 128, 64, 32};
    const int Cout_[6] = {32, 64, 128, 64, 32, 1};
    const int Win_[6]  = {511, 255, 128, 128, 128, 128};
    const int Wout_[6] = {255, 128, 128, 128, 128, 128};
    const int Str_[6]  = {2, 2, 1, 1, 1, 1};
    const int Pad_[6]  = {0, 1, 1, 1, 1, 1};
    float* pp[2] = {bufA, bufB};
    for (int l = 0; l < 6; ++l) {
        const float* cw  = (const float*)d_in[1 + l * 5 + 0];
        const float* cb  = (const float*)d_in[1 + l * 5 + 1];
        const float* ing = (const float*)d_in[1 + l * 5 + 2];
        const float* inb = (const float*)d_in[1 + l * 5 + 3];
        const float* pa  = (const float*)d_in[1 + l * 5 + 4];
        float* src = pp[l & 1];
        float* dst = pp[(l & 1) ^ 1];
        int KT = (Cin_[l] + 31) / 32;
        int Np = ((Cout_[l] + 15) / 16) * 16;
        {
            int tot = 3 * KT * Np * 32;
            k_packConvW<<<(tot + 255) / 256, 256, 0, stream>>>(cw, packC, Cin_[l], Cout_[l], KT, Np);
        }
        int tilesM = (256 * Wout_[l]) / 16;
        int tilesN = Np / 16;
        k_conv1x3<<<(unsigned)(BATCH * tilesM * tilesN), 32, 0, stream>>>(
            src, packC, cb, dst, Cin_[l], Cout_[l], 256, Win_[l], Wout_[l],
            Str_[l], Pad_[l], KT, Np);
        int HW = 256 * Wout_[l];
        int Cc = Cout_[l];
        k_zero<<<(BATCH * Cc * 2 + 255) / 256, 256, 0, stream>>>(stat, BATCH * Cc * 2);
        {
            size_t plane = (size_t)HW * Cc;
            unsigned nchunk = (unsigned)((plane + IN_CHUNK - 1) / IN_CHUNK);
            dim3 grid(nchunk, BATCH);
            k_instats<<<grid, 256, 0, stream>>>(dst, stat, Cc, HW);
            size_t tot = (size_t)BATCH * plane;
            k_inapply<<<(unsigned)((tot + 255) / 256), 256, 0, stream>>>(dst, stat, ing, inb, pa, Cc, HW);
        }
    }
    {
        int tot = BATCH * S_LEN * DMODEL;
        k_seq<<<(tot + 255) / 256, 256, 0, stream>>>(bufA, cls, xb);
    }

    // ---- mLSTM blocks ----
    for (int kblk = 0; kblk < 4; ++kblk) {
        int base = 32 + kblk * 15;
        const float* ln_g  = (const float*)d_in[base + 0];
        const float* ln_b  = (const float*)d_in[base + 1];
        const float* w_up  = (const float*)d_in[base + 2];
        const float* conv_w= (const float*)d_in[base + 3];
        const float* conv_b= (const float*)d_in[base + 4];
        const float* w_q   = (const float*)d_in[base + 5];
        const float* w_k   = (const float*)d_in[base + 6];
        const float* w_v   = (const float*)d_in[base + 7];
        const float* w_i   = (const float*)d_in[base + 8];
        const float* b_i   = (const float*)d_in[base + 9];
        const float* w_f   = (const float*)d_in[base + 10];
        const float* b_f   = (const float*)d_in[base + 11];
        const float* gn_g  = (const float*)d_in[base + 12];
        const float* skip  = (const float*)d_in[base + 13];
        const float* w_down= (const float*)d_in[base + 14];

        k_ln<<<(unsigned)TOK, DMODEL, 0, stream>>>(xb, ln_g, ln_b, xn);

        k_packW<<<(DMODEL * 512 + 255) / 256, 256, 0, stream>>>(w_up, packG, DMODEL, 512);
        k_gemm<<<(unsigned)((TOK / 16) * (512 / 64)), 128, 0, stream>>>(
            xn, packG, nullptr, nullptr, up, DMODEL, 512);
        {
            int tot = (int)(TOK * INNER_);
            k_dwconv<<<(tot + 255) / 256, 256, 0, stream>>>(up, conv_w, conv_b, ca, tot);
        }
        {
            int tot = (int)(TOK * 64);
            k_blocked<<<(tot + 255) / 256, 256, 0, stream>>>(ca, INNER_, w_q, qb, tot);
            k_blocked<<<(tot + 255) / 256, 256, 0, stream>>>(ca, INNER_, w_k, kb, tot);
            k_blocked<<<(tot + 255) / 256, 256, 0, stream>>>(up, 512,    w_v, vb, tot);
        }
        {
            int tot = (int)(TOK * NHEAD);
            k_gates<<<(tot + 255) / 256, 256, 0, stream>>>(qb, kb, vb, w_i, b_i, w_f, b_f,
                                                           igb, lfb, tot);
        }
        k_scan<<<1, 128, 0, stream>>>(igb, lfb, cfb, ubuf, abuf);
        {
            int tot = BATCH * NHEAD * DHEAD * SPAD;
            k_vtrans<<<(tot + 255) / 256, 256, 0, stream>>>(vb, vt);
        }
        k_attn<<<(unsigned)(BATCH * NHEAD * 17), 32, 0, stream>>>(qb, kb, vt, cfb, ubuf, abuf, hbuf);
        k_gnfuse<<<(unsigned)(TOK * NHEAD), DHEAD, 0, stream>>>(hbuf, up, ca, gn_g, skip, gin);

        k_packW<<<(INNER_ * DMODEL + 255) / 256, 256, 0, stream>>>(w_down, packG, INNER_, DMODEL);
        k_gemm<<<(unsigned)((TOK / 16) * (DMODEL / 64)), 128, 0, stream>>>(
            gin, packG, nullptr, xb, xb, INNER_, DMODEL);
    }

    k_head<<<BATCH, DMODEL, 0, stream>>>(xb, post_g, post_b, fc_lg, fc_lb, fc_w, fc_b,
                                         (float*)d_out);
}
